// SNNModel_11201274708706
// MI455X (gfx1250) — matrix operand order repaired and checked by reading, not compiled, not run on hardware
//
#include <hip/hip_runtime.h>
#include <stddef.h>
#include <stdint.h>

// ---------------------------------------------------------------------------
// Fused SNN forward for MI455X (gfx1250, wave32):
//   Phase 1: cur1 = x @ W1^T + b1 via v_wmma_f32_16x16x32_bf16.
//            x (205 MB) streamed once with NT b128 loads (~8.8 us HBM floor).
//            W1 staged to LDS as bf16 in double-buffered 8 KB k-chunks
//            (one barrier/iter; next chunk's global loads overlap WMMAs).
//            B fragments use a depth-2 register pipeline (bA/bB); the chunk
//            loop is kept rolled (#pragma unroll 1) so the accumulators stay
//            pinned and WMMAs issue in-place (no v_mov_b64 shuffling).
//   Phase 2: 25-step LIF recurrence entirely in VGPRs (cur1/mem1 resident),
//            per-step [128 -> 2] matvec via VALU + ds_swizzle butterfly.
// One wave owns a 16x128 tile; one 256-thread block owns 128 batch rows.
// ---------------------------------------------------------------------------

typedef __attribute__((ext_vector_type(8)))  float          v8f;
typedef __attribute__((ext_vector_type(16))) __bf16         v16bf;
typedef __attribute__((ext_vector_type(8)))  unsigned int   v8u;
typedef __attribute__((ext_vector_type(4)))  unsigned int   u4;
typedef __attribute__((ext_vector_type(4)))  float          f4;

constexpr int DIN = 784, DH = 128;
constexpr int NCHUNK = (DIN + 31) / 32;   // 25 (24 full + one 16-wide tail)
constexpr int LSTRIDE = 40;  // halves; 80 B = 16 B * 5 (odd) -> conflict-free

// Pack two f32 -> two bf16 in one dword (lo in [15:0], hi in [31:16]).
__device__ __forceinline__ unsigned int pack_bf16x2(float lo, float hi) {
#if __has_builtin(__builtin_amdgcn_cvt_pk_bf16_f32)
  typedef __attribute__((ext_vector_type(2))) __bf16 v2bf;
  v2bf r = __builtin_amdgcn_cvt_pk_bf16_f32(lo, hi);
  return __builtin_bit_cast(unsigned int, r);
#else
  // single v_perm_b32: take high 16 bits of each f32 (truncation)
  unsigned int lb = __builtin_bit_cast(unsigned int, lo);
  unsigned int hb = __builtin_bit_cast(unsigned int, hi);
  return __builtin_amdgcn_perm(hb, lb, 0x07060302u);
#endif
}

#define PACK8(dst, base, g0, g1)                   \
  dst[(base)+0] = pack_bf16x2((g0)[0], (g0)[1]);   \
  dst[(base)+1] = pack_bf16x2((g0)[2], (g0)[3]);   \
  dst[(base)+2] = pack_bf16x2((g1)[0], (g1)[1]);   \
  dst[(base)+3] = pack_bf16x2((g1)[2], (g1)[3]);

__device__ __forceinline__ v16bf load_bfrag(const unsigned short* bp) {
  const u4 q0 = ((const u4*)bp)[0];     // ds_load_b128
  const u4 q1 = ((const u4*)bp)[2];     // ds_load_b128
  v8u bu;
  bu[0] = q0[0]; bu[1] = q0[1]; bu[2] = q0[2]; bu[3] = q0[3];
  bu[4] = q1[0]; bu[5] = q1[1]; bu[6] = q1[2]; bu[7] = q1[3];
  return __builtin_bit_cast(v16bf, bu);
}

__device__ __forceinline__ v8f wmma_bf16(v16bf a, v16bf b, v8f c) {
  return __builtin_amdgcn_wmma_f32_16x16x32_bf16(
      false, a, false, b, (short)0, c, false, false);
}

template <bool FULL_A, bool DO_STAGE>
__device__ __forceinline__ void gemm_chunk(
    int I, v8f (&acc)[8], const float* xrow, const float* wsrc,
    unsigned short (&w1s)[2][128 * LSTRIDE],
    int hi8, int hi16, int nl, int trow, int thalf) {
  const int k0 = I * 32;
  __syncthreads();                       // chunk I staged; other buffer free

  // issue next-chunk W1 global loads first (fly during the WMMAs)
  f4 g0 = {}, g1 = {}, g2 = {}, g3 = {};
  const int kb_n = k0 + 32 + thalf * 16;
  if (DO_STAGE && kb_n + 16 <= DIN) {
    g0 = *(const f4*)(wsrc + kb_n + 0);
    g1 = *(const f4*)(wsrc + kb_n + 4);
    g2 = *(const f4*)(wsrc + kb_n + 8);
    g3 = *(const f4*)(wsrc + kb_n + 12);
  }
  if (FULL_A && k0 + 64 < DIN)
    __builtin_prefetch(xrow + k0 + 64 + hi8, 0, 0);

  // A fragment (16x32): lane holds K = k0+hi8+{0..7} and k0+16+hi8+{0..7}
  v8u au = {};
  {
    const f4 f0 = __builtin_nontemporal_load((const f4*)(xrow + k0 + hi8));
    const f4 f1 = __builtin_nontemporal_load((const f4*)(xrow + k0 + hi8 + 4));
    PACK8(au, 0, f0, f1)
  }
  if (FULL_A) {                          // tail chunk: uniform zero-pad
    const f4 f0 = __builtin_nontemporal_load((const f4*)(xrow + k0 + 16 + hi8));
    const f4 f1 = __builtin_nontemporal_load((const f4*)(xrow + k0 + 16 + hi8 + 4));
    PACK8(au, 4, f0, f1)
  }
  const v16bf a = __builtin_bit_cast(v16bf, au);

  // depth-2 B-fragment pipeline: 2 x ds_load_b128 per fragment from LDS
  const unsigned short* base = &w1s[I & 1][nl * LSTRIDE + hi8];
  v16bf bA = load_bfrag(base + 0 * 16 * LSTRIDE);
  v16bf bB = load_bfrag(base + 1 * 16 * LSTRIDE);
  acc[0] = wmma_bf16(a, bA, acc[0]);  bA = load_bfrag(base + 2 * 16 * LSTRIDE);
  acc[1] = wmma_bf16(a, bB, acc[1]);  bB = load_bfrag(base + 3 * 16 * LSTRIDE);
  acc[2] = wmma_bf16(a, bA, acc[2]);  bA = load_bfrag(base + 4 * 16 * LSTRIDE);
  acc[3] = wmma_bf16(a, bB, acc[3]);  bB = load_bfrag(base + 5 * 16 * LSTRIDE);
  acc[4] = wmma_bf16(a, bA, acc[4]);  bA = load_bfrag(base + 6 * 16 * LSTRIDE);
  acc[5] = wmma_bf16(a, bB, acc[5]);  bB = load_bfrag(base + 7 * 16 * LSTRIDE);
  acc[6] = wmma_bf16(a, bA, acc[6]);
  acc[7] = wmma_bf16(a, bB, acc[7]);

  // finish staging: pack + ds_store (after this chunk's ds_loads; LDS is
  // in-order per wave, so storing late keeps fragment reads unstalled)
  if (DO_STAGE) {
    u4 s0 = {}, s1 = {};
    if (kb_n + 16 <= DIN) { PACK8(s0, 0, g0, g1) PACK8(s1, 0, g2, g3) }
    unsigned short* dst = &w1s[(I + 1) & 1][trow * LSTRIDE + thalf * 16];
    ((u4*)dst)[0] = s0;
    ((u4*)dst)[1] = s1;
  }
}

__global__ __launch_bounds__(256)
void snn_fused_kernel(const float* __restrict__ x,  const float* __restrict__ W1,
                      const float* __restrict__ b1, const float* __restrict__ W2,
                      const float* __restrict__ b2, const int* __restrict__ nsteps,
                      float* __restrict__ out, int B) {
  __shared__ __align__(16) unsigned short w1s[2][128 * LSTRIDE];   // 2 x 10 KB

  const int lane = threadIdx.x & 31;
  const int wave = threadIdx.x >> 5;
  const int nl   = lane & 15;            // C-layout: N = nt*16 + nl
  const int hi   = lane >> 4;            // C-layout: M = j + 8*hi
  const int hi8  = hi * 8, hi16 = hi * 16;
  const int m0   = blockIdx.x * 128 + wave * 16;

  int rowA = m0 + nl;                    // A-layout: M = nl
  if (rowA > B - 1) rowA = B - 1;
  if (rowA < 0) rowA = 0;
  const float* xrow = x + (size_t)rowA * DIN;

  // staging assignment: thread -> (row, k-half) of a 128x32 W1 chunk
  const int trow  = threadIdx.x >> 1;    // 0..127
  const int thalf = threadIdx.x & 1;     // 0..1
  const float* wsrc = W1 + (size_t)trow * DIN;

  // ---------------- Phase 1: GEMM1 (WMMA bf16, f32 accum) ----------------
  v8f acc[8] = {};   // acc[nt][j] = cur1[m0 + j + 8*hi][nt*16 + nl]

  // prologue: stage chunk 0 into buffer 0 (k = 0..31, always in range)
  {
    const int kb = thalf * 16;
    const f4 g0 = *(const f4*)(wsrc + kb + 0);
    const f4 g1 = *(const f4*)(wsrc + kb + 4);
    const f4 g2 = *(const f4*)(wsrc + kb + 8);
    const f4 g3 = *(const f4*)(wsrc + kb + 12);
    u4 s0, s1;
    PACK8(s0, 0, g0, g1)
    PACK8(s1, 0, g2, g3)
    unsigned short* dst = &w1s[0][trow * LSTRIDE + thalf * 16];
    ((u4*)dst)[0] = s0;
    ((u4*)dst)[1] = s1;
  }

  // keep rolled: in-loop software pipeline already covers latency; unrolling
  // only forces accumulator copies at the unroll seams (seen in round 3/4 asm)
  #pragma unroll 1
  for (int i = 0; i < NCHUNK - 1; ++i)
    gemm_chunk<true, true>(i, acc, xrow, wsrc, w1s, hi8, hi16, nl, trow, thalf);
  gemm_chunk<false, false>(NCHUNK - 1, acc, xrow, wsrc, w1s,
                           hi8, hi16, nl, trow, thalf);   // K 784..799 zero

  // bias b1 (per lane: column n = nt*16+nl, same for all 8 rows)
  #pragma unroll
  for (int nt = 0; nt < 8; ++nt) {
    const float bb = b1[nt * 16 + nl];
    #pragma unroll
    for (int j = 0; j < 8; ++j) acc[nt][j] += bb;
  }

  // ---------------- Phase 2: LIF recurrence (all in registers) -----------
  float w20[8], w21[8];                 // W2[o][nt*16+nl]
  #pragma unroll
  for (int nt = 0; nt < 8; ++nt) {
    w20[nt] = W2[nt * 16 + nl];
    w21[nt] = W2[DH + nt * 16 + nl];
  }
  const float bb0 = b2[0], bb1 = b2[1];

  float mem1[8][8];
  #pragma unroll
  for (int nt = 0; nt < 8; ++nt)
    #pragma unroll
    for (int j = 0; j < 8; ++j) mem1[nt][j] = 0.0f;

  float m2a[8], m2b[8];                 // mem2 for rows j (this half), o = 0/1
  #pragma unroll
  for (int j = 0; j < 8; ++j) { m2a[j] = 0.0f; m2b[j] = 0.0f; }

  unsigned long long prev1 = 0ull;      // spk1 history bitmask (64 elems/lane)
  unsigned int       prev2 = 0u;        // spk2 history bitmask (16 bits)

  const int T = nsteps[0];
  const long long tstride = (long long)B * 2;

  #pragma unroll 1
  for (int t = 0; t < T; ++t) {
    float p0[8], p1[8];
    #pragma unroll
    for (int j = 0; j < 8; ++j) { p0[j] = 0.0f; p1[j] = 0.0f; }
    unsigned long long nb = 0ull;

    #pragma unroll
    for (int nt = 0; nt < 8; ++nt) {
      #pragma unroll
      for (int j = 0; j < 8; ++j) {
        // mem1 = beta*mem1 + cur1 - thr*spk_prev ; spk = (mem1 - thr) > 0
        float m = fmaf(0.9f, mem1[nt][j], acc[nt][j]);
        m -= (float)((prev1 >> (nt * 8 + j)) & 1ull);
        mem1[nt][j] = m;
        const bool fire = m > 1.0f;
        const float s = fire ? 1.0f : 0.0f;
        nb |= ((unsigned long long)fire) << (nt * 8 + j);
        p0[j] = fmaf(s, w20[nt], p0[j]);    // partial cur2, o=0
        p1[j] = fmaf(s, w21[nt], p1[j]);    // partial cur2, o=1
      }
    }
    prev1 = nb;

    // butterfly reduce over the 16 lanes of each half (xor 1,2,4,8 -> ds ops)
    #pragma unroll
    for (int j = 0; j < 8; ++j) {
      float a0 = p0[j], a1 = p1[j];
      #pragma unroll
      for (int msk = 1; msk < 16; msk <<= 1) {
        a0 += __shfl_xor(a0, msk, 32);
        a1 += __shfl_xor(a1, msk, 32);
      }
      p0[j] = a0; p1[j] = a1;
    }

    unsigned int nb2 = 0u;
    #pragma unroll
    for (int j = 0; j < 8; ++j) {
      float u0 = fmaf(0.9f, m2a[j], p0[j] + bb0) - (float)((prev2 >> (2*j  )) & 1u);
      float u1 = fmaf(0.9f, m2b[j], p1[j] + bb1) - (float)((prev2 >> (2*j+1)) & 1u);
      m2a[j] = u0; m2b[j] = u1;
      const bool f0b = u0 > 1.0f, f1b = u1 > 1.0f;
      nb2 |= (f0b ? 1u : 0u) << (2*j);
      nb2 |= (f1b ? 1u : 0u) << (2*j+1);
      const int row = m0 + hi8 + j;
      if (nl == 0 && row < B) {           // values replicated across the half
        float* o = out + (long long)t * tstride + (long long)row * 2;
        __builtin_nontemporal_store(f0b ? 1.0f : 0.0f, o + 0);
        __builtin_nontemporal_store(f1b ? 1.0f : 0.0f, o + 1);
      }
    }
    prev2 = nb2;
  }
}

extern "C" void kernel_launch(void* const* d_in, const int* in_sizes, int n_in,
                              void* d_out, int out_size, void* d_ws, size_t ws_size,
                              hipStream_t stream) {
  const float* x  = (const float*)d_in[0];
  const float* W1 = (const float*)d_in[1];
  const float* b1 = (const float*)d_in[2];
  const float* W2 = (const float*)d_in[3];
  const float* b2 = (const float*)d_in[4];
  const int* nst  = (const int*)d_in[5];
  float* out = (float*)d_out;

  const int B = in_sizes[0] / 784;           // 65536
  const int blocks = (B + 127) / 128;        // 512 blocks x 8 waves x 16 rows
  snn_fused_kernel<<<blocks, 256, 0, stream>>>(x, W1, b1, W2, b2, nst, out, B);
  (void)n_in; (void)out_size; (void)d_ws; (void)ws_size;
}
